// TransformerBlock_35888746725805
// MI455X (gfx1250) — compile-verified, matrix operand order checked by reading, not run on hardware
//
#include <hip/hip_runtime.h>

// ---------------- problem constants ----------------
constexpr int B_    = 4;
constexpr int SEQ_  = 2048;
constexpr int DIM_  = 2048;
constexpr int NH_   = 16;
constexpr int HD_   = 128;
constexpr int HID_  = 5632;
constexpr int NTOK_ = B_ * SEQ_;   // 8192
constexpr float EPS_ = 1e-5f;

// ---------------- types ----------------
typedef __attribute__((ext_vector_type(16))) __bf16 v16bf;
typedef __attribute__((ext_vector_type(8)))  float  v8f;
typedef __attribute__((ext_vector_type(8)))  unsigned int u32x8;

__device__ __forceinline__ unsigned short f2bf(float f) {
  unsigned int u = __float_as_uint(f);
  unsigned int r = u + 0x7FFFu + ((u >> 16) & 1u);
  return (unsigned short)(r >> 16);
}

// Load a 16x32 bf16 A-fragment (or symmetric B-fragment from an N-major
// matrix) per CDNA5 ISA 7.12.2:
//  lanes 0-15 : row r, K = {0..7} then {16..23}
//  lanes16-31 : row r, K = {8..15} then {24..31}
// Two contiguous 16B loads per lane.
__device__ __forceinline__ v16bf frag_ld(const unsigned short* __restrict__ p, int ld) {
  const int lane = threadIdx.x & 31;
  const int r = lane & 15, h = lane >> 4;
  const unsigned short* a = p + (size_t)r * ld + (h << 3);
  const uint4 lo = *(const uint4*)(a);
  const uint4 hi = *(const uint4*)(a + 16);
  u32x8 u;
  u[0] = lo.x; u[1] = lo.y; u[2] = lo.z; u[3] = lo.w;
  u[4] = hi.x; u[5] = hi.y; u[6] = hi.z; u[7] = hi.w;
  return __builtin_bit_cast(v16bf, u);
}

__device__ __forceinline__ v8f wmma_bf16(v16bf a, v16bf b, v8f c) {
  return __builtin_amdgcn_wmma_f32_16x16x32_bf16(false, a, false, b, (short)0, c, false, false);
}

// ---- CDNA5 async global->LDS copy (ASYNCcnt-tracked, no VGPR staging) ----
// LDS address for a __shared__ object is the low 32 bits of its generic
// address (ISA 10.2).
__device__ __forceinline__ void async_ld_b128(const void* gptr, void* lptr) {
  unsigned lds = (unsigned)(size_t)lptr;
  unsigned long long ga = (unsigned long long)(size_t)gptr;
  asm volatile("global_load_async_to_lds_b128 %0, %1, off"
               :: "v"(lds), "v"(ga) : "memory");
}
template <int N>
__device__ __forceinline__ void wait_async() {
  asm volatile("s_wait_asynccnt %0" :: "n"(N) : "memory");
}

// Stage a 128-row x 32-col bf16 tile (row stride ldK elems) into LDS
// (row stride 32 elems). 512 x 16B chunks over 256 threads -> 2 async
// instructions per wave (per-wave ASYNCcnt += 2).
__device__ __forceinline__ void stage_tile(const unsigned short* __restrict__ g,
                                           int ldK, unsigned short* l) {
  const int t = threadIdx.x;
  #pragma unroll
  for (int c = t; c < 512; c += 256) {
    const int row = c >> 2;
    const int co  = (c & 3) * 8;  // elements (16B)
    async_ld_b128(g + (size_t)row * ldK + co, l + row * 32 + co);
  }
}

// ---------------- weight convert + transpose (f32 [R,C] -> bf16 [C,R]) ----
__global__ void convT_kernel(const float* __restrict__ IN, unsigned short* __restrict__ OUT,
                             int Rows, int Cols) {
  size_t idx = (size_t)blockIdx.x * blockDim.x + threadIdx.x;
  size_t tot = (size_t)Rows * Cols;
  if (idx >= tot) return;
  size_t rr = idx / Cols;
  int    c  = (int)(idx - rr * Cols);
  OUT[(size_t)c * Rows + rr] = f2bf(IN[idx]);
}

// ---------------- RMSNorm (f32 -> bf16), one block per row ----------------
__global__ __launch_bounds__(256) void rmsnorm_kernel(const float* __restrict__ X,
                                                      const float* __restrict__ W,
                                                      unsigned short* __restrict__ OUT, int D) {
  __shared__ float red[8];
  const int row = blockIdx.x;
  const float* x = X + (size_t)row * D;
  float ss = 0.f;
  for (int i = threadIdx.x; i < D; i += 256) { float v = x[i]; ss += v * v; }
  for (int off = 1; off < 32; off <<= 1) ss += __shfl_xor(ss, off, 32);
  if ((threadIdx.x & 31) == 0) red[threadIdx.x >> 5] = ss;
  __syncthreads();
  float tot = 0.f;
  #pragma unroll
  for (int i = 0; i < 8; i++) tot += red[i];
  const float rs = rsqrtf(tot / (float)D + EPS_);
  for (int i = threadIdx.x; i < D; i += 256)
    OUT[(size_t)row * D + i] = f2bf(x[i] * rs * W[i]);
}

// ---------------- RoPE: f32 [NTOK,DIM] -> bf16, interleaved pairs ---------
__global__ void rope_kernel(const float* __restrict__ X, const float* __restrict__ C,
                            const float* __restrict__ S, unsigned short* __restrict__ OUT) {
  size_t idx = (size_t)blockIdx.x * blockDim.x + threadIdx.x;  // NTOK*NH*64
  if (idx >= (size_t)NTOK_ * NH_ * 64) return;
  int j = (int)(idx & 63);
  int h = (int)((idx >> 6) & 15);
  size_t tok = idx >> 10;
  int s = (int)(tok & (SEQ_ - 1));
  size_t base = tok * DIM_ + (size_t)h * HD_ + 2 * j;
  float a = X[base], b = X[base + 1];
  float c = C[(size_t)s * 64 + j], sn = S[(size_t)s * 64 + j];
  OUT[base]     = f2bf(a * c - b * sn);
  OUT[base + 1] = f2bf(a * sn + b * c);
}

// ---------------- V transpose: [tok, h*HD+d] -> [(b,h,d), s] --------------
__global__ void vtrans_kernel(const unsigned short* __restrict__ V,
                              unsigned short* __restrict__ VT) {
  size_t idx = (size_t)blockIdx.x * blockDim.x + threadIdx.x;  // B*NH*HD*SEQ
  if (idx >= (size_t)B_ * NH_ * HD_ * SEQ_) return;
  int s = (int)(idx & (SEQ_ - 1));
  int d = (int)((idx >> 11) & (HD_ - 1));
  int h = (int)((idx >> 18) & 15);
  int b = (int)(idx >> 22);
  VT[idx] = V[((size_t)b * SEQ_ + s) * DIM_ + (size_t)h * HD_ + d];
}

// ---------------- WMMA GEMM with async-LDS double buffering ---------------
// A  : [M,K] bf16 row-major; Bt : [N,K] bf16 row-major (B transposed).
// Block = 128x128, 8 waves (4x2), wave tile = 32x64 (2x4 subtiles of 16x16).
// A/B k-slices (128x32) are staged into LDS with GLOBAL_LOAD_ASYNC_TO_LDS,
// double-buffered: prefetch tile k+1, s_wait_asynccnt <= just-issued, barrier,
// compute tile k from LDS.
enum { EPI_BF16 = 0, EPI_F32 = 1, EPI_ADDRES_F32 = 2, EPI_SILU_BF16 = 3 };

template <int EPI>
__global__ __launch_bounds__(256) void gemm_wmma_kernel(
    const unsigned short* __restrict__ A, const unsigned short* __restrict__ Bt,
    const unsigned short* __restrict__ B2t, const float* __restrict__ RES,
    void* __restrict__ OUT, int M, int N, int K) {
  __shared__ __align__(16) unsigned short sA[2][128 * 32];
  __shared__ __align__(16) unsigned short sB[2][128 * 32];
  __shared__ __align__(16) unsigned short sB2[2][128 * 32];

  const int wv = threadIdx.x >> 5;
  const int lane = threadIdx.x & 31;
  const int r = lane & 15, hh = lane >> 4;
  const int mloc = (wv >> 1) * 32;         // wave tile inside block
  const int nloc = (wv & 1) * 64;
  const int m0 = blockIdx.y * 128 + mloc;
  const int n0 = blockIdx.x * 128 + nloc;
  const unsigned short* Ab = A  + (size_t)blockIdx.y * 128 * K;
  const unsigned short* Bb = Bt + (size_t)blockIdx.x * 128 * K;
  const unsigned short* B2b = (EPI == EPI_SILU_BF16) ? (B2t + (size_t)blockIdx.x * 128 * K) : nullptr;

  const v8f vzero = {0.f, 0.f, 0.f, 0.f, 0.f, 0.f, 0.f, 0.f};
  v8f acc[2][4], acc2[2][4];
  #pragma unroll
  for (int i = 0; i < 2; i++)
    #pragma unroll
    for (int j = 0; j < 4; j++) { acc[i][j] = vzero; acc2[i][j] = vzero; }

  // prefetch tile 0
  stage_tile(Ab, K, sA[0]);
  stage_tile(Bb, K, sB[0]);
  if (EPI == EPI_SILU_BF16) stage_tile(B2b, K, sB2[0]);

  const int nk = K / 32;
  for (int kt = 0; kt < nk; ++kt) {
    const int cur = kt & 1, nxt = cur ^ 1;
    if (kt + 1 < nk) {
      stage_tile(Ab + (size_t)(kt + 1) * 32, K, sA[nxt]);
      stage_tile(Bb + (size_t)(kt + 1) * 32, K, sB[nxt]);
      if (EPI == EPI_SILU_BF16) stage_tile(B2b + (size_t)(kt + 1) * 32, K, sB2[nxt]);
      if (EPI == EPI_SILU_BF16) wait_async<6>(); else wait_async<4>();
    } else {
      wait_async<0>();
    }
    __syncthreads();   // every wave's portion of tile `cur` has landed in LDS

    v16bf a0 = frag_ld(sA[cur] + (size_t)mloc * 32, 32);
    v16bf a1 = frag_ld(sA[cur] + (size_t)(mloc + 16) * 32, 32);
    #pragma unroll
    for (int j = 0; j < 4; j++) {
      v16bf b = frag_ld(sB[cur] + (size_t)(nloc + j * 16) * 32, 32);
      acc[0][j] = wmma_bf16(a0, b, acc[0][j]);
      acc[1][j] = wmma_bf16(a1, b, acc[1][j]);
      if (EPI == EPI_SILU_BF16) {
        v16bf b2 = frag_ld(sB2[cur] + (size_t)(nloc + j * 16) * 32, 32);
        acc2[0][j] = wmma_bf16(a0, b2, acc2[0][j]);
        acc2[1][j] = wmma_bf16(a1, b2, acc2[1][j]);
      }
    }
    __syncthreads();   // safe to overwrite buffer `cur` with tile kt+2
  }

  #pragma unroll
  for (int i = 0; i < 2; i++)
    #pragma unroll
    for (int j = 0; j < 4; j++)
      #pragma unroll
      for (int e = 0; e < 8; e++) {
        int row = m0 + i * 16 + hh * 8 + e;
        int col = n0 + j * 16 + r;
        size_t idx = (size_t)row * N + col;
        float v = acc[i][j][e];
        if (EPI == EPI_BF16) {
          ((unsigned short*)OUT)[idx] = f2bf(v);
        } else if (EPI == EPI_F32) {
          ((float*)OUT)[idx] = v;
        } else if (EPI == EPI_ADDRES_F32) {
          ((float*)OUT)[idx] = v + RES[idx];
        } else {  // SiLU(A@B) * (A@B2) -> bf16
          float g = v;
          float sg = g / (1.f + __expf(-g));
          ((unsigned short*)OUT)[idx] = f2bf(sg * acc2[i][j][e]);
        }
      }
}

// ---------------- Flash attention ----------------------------------------
// grid = (SEQ/128, B*NH), block = 256 (8 waves). Each wave owns 16 q rows
// and the full HD=128 output, streaming causal kv tiles of 32.
__global__ __launch_bounds__(256) void attn_kernel(const unsigned short* __restrict__ Q,
                                                   const unsigned short* __restrict__ Kb,
                                                   const unsigned short* __restrict__ Vt,
                                                   unsigned short* __restrict__ O) {
  __shared__ __align__(16) unsigned short pl[8 * 16 * 32];  // per-wave 16x32 P tile
  const int wv = threadIdx.x >> 5;
  const int lane = threadIdx.x & 31;
  const int r = lane & 15, hh = lane >> 4;
  const int bh = blockIdx.y;
  const int b = bh >> 4, h = bh & 15;
  const int q0 = blockIdx.x * 128 + wv * 16;          // seq position of first q row
  const size_t tok0 = (size_t)b * SEQ_ + q0;
  const size_t qbase = tok0 * DIM_ + (size_t)h * HD_;
  const float scale = 0.088388347648318447f;          // 1/sqrt(128)
  const v8f vzero = {0.f, 0.f, 0.f, 0.f, 0.f, 0.f, 0.f, 0.f};

  v16bf qa[4];
  #pragma unroll
  for (int t = 0; t < 4; t++) qa[t] = frag_ld(Q + qbase + t * 32, DIM_);

  v8f oacc[8];
  float mr[8], lr[8];
  #pragma unroll
  for (int t = 0; t < 8; t++) oacc[t] = vzero;
  #pragma unroll
  for (int e = 0; e < 8; e++) { mr[e] = -3.0e38f; lr[e] = 0.f; }

  const int ntiles = (q0 + 16 + 31) / 32;  // causal bound for this wave
  for (int kt = 0; kt < ntiles; ++kt) {
    const int kv0 = kt * 32;
    v8f s0 = vzero, s1 = vzero;
    const size_t kbase = ((size_t)b * SEQ_ + kv0) * DIM_ + (size_t)h * HD_;
    #pragma unroll
    for (int t = 0; t < 4; t++) {
      v16bf k0f = frag_ld(Kb + kbase + t * 32, DIM_);
      v16bf k1f = frag_ld(Kb + kbase + (size_t)16 * DIM_ + t * 32, DIM_);
      s0 = wmma_bf16(qa[t], k0f, s0);
      s1 = wmma_bf16(qa[t], k1f, s1);
    }
    // online softmax per row (rows live as element index e, cols across lanes)
    #pragma unroll
    for (int e = 0; e < 8; e++) {
      const int qi = q0 + hh * 8 + e;
      float v0 = s0[e] * scale, v1 = s1[e] * scale;
      if (kv0 + r > qi)      v0 = -3.0e38f;
      if (kv0 + 16 + r > qi) v1 = -3.0e38f;
      float rm = fmaxf(v0, v1);
      for (int off = 1; off < 16; off <<= 1) rm = fmaxf(rm, __shfl_xor(rm, off, 32));
      const float mn = fmaxf(mr[e], rm);
      const float alpha = __expf(mr[e] - mn);
      const float e0 = __expf(v0 - mn), e1 = __expf(v1 - mn);
      float rs = e0 + e1;
      for (int off = 1; off < 16; off <<= 1) rs += __shfl_xor(rs, off, 32);
      lr[e] = lr[e] * alpha + rs;
      mr[e] = mn;
      #pragma unroll
      for (int t = 0; t < 8; t++) oacc[t][e] *= alpha;
      // stage P into this wave's LDS tile (row-major 16x32 bf16)
      pl[wv * 512 + (hh * 8 + e) * 32 + r]      = f2bf(e0);
      pl[wv * 512 + (hh * 8 + e) * 32 + 16 + r] = f2bf(e1);
    }
    asm volatile("s_wait_dscnt 0" ::: "memory");  // wave-local LDS RAW
    v16bf pa = frag_ld(pl + wv * 512, 32);
    const size_t vbase = ((size_t)bh * HD_) * SEQ_ + kv0;
    #pragma unroll
    for (int t = 0; t < 8; t++) {
      v16bf vb = frag_ld(Vt + vbase + (size_t)(t * 16) * SEQ_, SEQ_);
      oacc[t] = wmma_bf16(pa, vb, oacc[t]);
    }
  }
  // write O (bf16) at [token, h*HD + d]
  #pragma unroll
  for (int t = 0; t < 8; t++)
    #pragma unroll
    for (int e = 0; e < 8; e++) {
      const int row = hh * 8 + e;
      const float val = oacc[t][e] / lr[e];
      O[(tok0 + row) * DIM_ + (size_t)h * HD_ + t * 16 + r] = f2bf(val);
    }
}

// ---------------- host launcher ------------------------------------------
extern "C" void kernel_launch(void* const* d_in, const int* in_sizes, int n_in,
                              void* d_out, int out_size, void* d_ws, size_t ws_size,
                              hipStream_t stream) {
  (void)in_sizes; (void)n_in; (void)out_size; (void)ws_size;
  const float* x   = (const float*)d_in[0];
  const float* wq  = (const float*)d_in[1];
  const float* wk  = (const float*)d_in[2];
  const float* wv  = (const float*)d_in[3];
  const float* wo  = (const float*)d_in[4];
  const float* w1  = (const float*)d_in[5];
  const float* w2  = (const float*)d_in[6];
  const float* w3  = (const float*)d_in[7];
  const float* anw = (const float*)d_in[8];
  const float* fnw = (const float*)d_in[9];
  const float* fc  = (const float*)d_in[10];
  const float* fs  = (const float*)d_in[11];
  float* out = (float*)d_out;

  char* ws = (char*)d_ws;
  size_t off = 0;
  auto take = [&](size_t bytes) -> char* {
    char* p = ws + off;
    off += (bytes + 255) & ~(size_t)255;
    return p;
  };
  unsigned short* bWq  = (unsigned short*)take((size_t)DIM_ * DIM_ * 2);
  unsigned short* bWk  = (unsigned short*)take((size_t)DIM_ * DIM_ * 2);
  unsigned short* bWv  = (unsigned short*)take((size_t)DIM_ * DIM_ * 2);
  unsigned short* bWo  = (unsigned short*)take((size_t)DIM_ * DIM_ * 2);
  unsigned short* bW1t = (unsigned short*)take((size_t)DIM_ * HID_ * 2);
  unsigned short* bW3t = (unsigned short*)take((size_t)DIM_ * HID_ * 2);
  unsigned short* bW2t = (unsigned short*)take((size_t)HID_ * DIM_ * 2);
  unsigned short* hn   = (unsigned short*)take((size_t)NTOK_ * DIM_ * 2);  // reused as fn
  float*          qf   = (float*)take((size_t)NTOK_ * DIM_ * 4);           // reused as h
  float*          kf   = (float*)take((size_t)NTOK_ * DIM_ * 4);
  unsigned short* qb   = (unsigned short*)take((size_t)NTOK_ * DIM_ * 2);
  unsigned short* kb   = (unsigned short*)take((size_t)NTOK_ * DIM_ * 2);
  unsigned short* vb   = (unsigned short*)take((size_t)NTOK_ * DIM_ * 2);
  unsigned short* vt   = (unsigned short*)take((size_t)NTOK_ * DIM_ * 2);
  unsigned short* ao   = (unsigned short*)take((size_t)NTOK_ * DIM_ * 2);
  unsigned short* act  = (unsigned short*)take((size_t)NTOK_ * HID_ * 2);

  const int thr = 256;
  auto blks = [&](size_t n) { return (unsigned)((n + thr - 1) / thr); };

  // 1) weights -> bf16, transposed to N-major
  convT_kernel<<<blks((size_t)DIM_ * DIM_), thr, 0, stream>>>(wq, bWq, DIM_, DIM_);
  convT_kernel<<<blks((size_t)DIM_ * DIM_), thr, 0, stream>>>(wk, bWk, DIM_, DIM_);
  convT_kernel<<<blks((size_t)DIM_ * DIM_), thr, 0, stream>>>(wv, bWv, DIM_, DIM_);
  convT_kernel<<<blks((size_t)DIM_ * DIM_), thr, 0, stream>>>(wo, bWo, DIM_, DIM_);
  convT_kernel<<<blks((size_t)DIM_ * HID_), thr, 0, stream>>>(w1, bW1t, DIM_, HID_);
  convT_kernel<<<blks((size_t)DIM_ * HID_), thr, 0, stream>>>(w3, bW3t, DIM_, HID_);
  convT_kernel<<<blks((size_t)HID_ * DIM_), thr, 0, stream>>>(w2, bW2t, HID_, DIM_);

  // 2) attn rmsnorm
  rmsnorm_kernel<<<NTOK_, thr, 0, stream>>>(x, anw, hn, DIM_);

  // 3) q/k/v projections
  dim3 gD(DIM_ / 128, NTOK_ / 128);
  gemm_wmma_kernel<EPI_F32><<<gD, thr, 0, stream>>>(hn, bWq, nullptr, nullptr, qf, NTOK_, DIM_, DIM_);
  gemm_wmma_kernel<EPI_F32><<<gD, thr, 0, stream>>>(hn, bWk, nullptr, nullptr, kf, NTOK_, DIM_, DIM_);
  gemm_wmma_kernel<EPI_BF16><<<gD, thr, 0, stream>>>(hn, bWv, nullptr, nullptr, vb, NTOK_, DIM_, DIM_);

  // 4) rope + V transpose
  rope_kernel<<<blks((size_t)NTOK_ * NH_ * 64), thr, 0, stream>>>(qf, fc, fs, qb);
  rope_kernel<<<blks((size_t)NTOK_ * NH_ * 64), thr, 0, stream>>>(kf, fc, fs, kb);
  vtrans_kernel<<<blks((size_t)NTOK_ * DIM_), thr, 0, stream>>>(vb, vt);

  // 5) flash attention
  attn_kernel<<<dim3(SEQ_ / 128, B_ * NH_), thr, 0, stream>>>(qb, kb, vt, ao);

  // 6) output projection + residual (h lives in qf)
  float* hbuf = qf;
  gemm_wmma_kernel<EPI_ADDRES_F32><<<gD, thr, 0, stream>>>(ao, bWo, nullptr, x, hbuf, NTOK_, DIM_, DIM_);

  // 7) ffn rmsnorm (reuse hn buffer)
  rmsnorm_kernel<<<NTOK_, thr, 0, stream>>>(hbuf, fnw, hn, DIM_);

  // 8) fused gate: silu(fn@w1) * (fn@w3) -> bf16
  dim3 gH(HID_ / 128, NTOK_ / 128);
  gemm_wmma_kernel<EPI_SILU_BF16><<<gH, thr, 0, stream>>>(hn, bW1t, bW3t, nullptr, act, NTOK_, HID_, DIM_);

  // 9) down projection + residual -> out (f32)
  gemm_wmma_kernel<EPI_ADDRES_F32><<<gD, thr, 0, stream>>>(act, bW2t, nullptr, hbuf, out, NTOK_, DIM_, HID_);
}
